// KVCacheAttention_55387898249691
// MI455X (gfx1250) — compile-verified, matrix operand order checked by reading, not run on hardware
//
#include <hip/hip_runtime.h>
#include <hip/hip_bf16.h>

typedef unsigned int u32;
typedef unsigned short u16;
typedef unsigned long long u64;
typedef __attribute__((ext_vector_type(4)))  u32    sg4;
typedef __attribute__((ext_vector_type(8)))  u32    sg8;
typedef __attribute__((ext_vector_type(8)))  float  v8f;
typedef __attribute__((ext_vector_type(8)))  u32    v8u;
typedef __attribute__((ext_vector_type(16))) __bf16 v16bf;

#define TQ   2048
#define HIDW 4096
#define NH   32
#define NKV  8
#define DHD  128
#define TMAXC 4096
#define ROPE_BASE 500000.0f

__device__ __forceinline__ u16 f2bf(float f) {
  u32 u = __builtin_bit_cast(u32, f);
  u32 r = u + 0x7FFFu + ((u >> 16) & 1u);
  return (u16)(r >> 16);
}
__device__ __forceinline__ float bf2f(u16 h) {
  u32 u = ((u32)h) << 16;
  return __builtin_bit_cast(float, u);
}
__device__ __forceinline__ v8f wmma_bf16(v8u a, v8u b, v8f c) {
  return __builtin_amdgcn_wmma_f32_16x16x32_bf16(
      false, __builtin_bit_cast(v16bf, a),
      false, __builtin_bit_cast(v16bf, b),
      (short)0, c, false, false);
}

// ---------------------------------------------------------------- TDM helper
// 2-D tensor_load_to_lds (VIMAGE op 196), D# per CDNA5 ISA.
// data_size = 2 bytes; groups 2/3 NULL (2-D). Values must be wave-uniform.
// pad_icode: LDS padding interval (0=2 dwords,1=4,...,7=256)
// pad_acode: LDS padding amount   (0=1 dword, ...,127=128)
__device__ __forceinline__ void tdm_load_2d_bf16(const void* gaddr, u32 lds_off,
                                                 u32 tdim0, u32 tdim1,
                                                 u32 tile0, u32 tile1, u32 stride0,
                                                 u32 pad_icode, u32 pad_acode) {
  u64 ga = (u64)(size_t)gaddr;
  sg4 g0;
  g0[0] = 1u;                                               // count=1, user mode
  g0[1] = lds_off;                                          // LDS byte address
  g0[2] = (u32)ga;                                          // global_addr[31:0]
  g0[3] = ((u32)(ga >> 32) & 0x01FFFFFFu) | 0x80000000u;    // addr[56:32] | type=2
  sg8 g1;
  g1[0] = (1u << 16) | (1u << 20) | (pad_icode << 22) | (pad_acode << 25);
  g1[1] = (tdim0 & 0xFFFFu) << 16;                          // tensor_dim0[15:0]
  g1[2] = (tdim0 >> 16) | ((tdim1 & 0xFFFFu) << 16);        // dim0 hi | dim1 lo
  g1[3] = (tdim1 >> 16) | (tile0 << 16);                    // dim1 hi | tile_dim0
  g1[4] = tile1;                                            // tile_dim1 (tile_dim2=0)
  g1[5] = stride0;                                          // tensor_dim0_stride lo
  g1[6] = 0u;
  g1[7] = 0u;
  asm volatile("tensor_load_to_lds %0, %1" ::"s"(g0), "s"(g1) : "memory");
}

// Load one 16x32 bf16 fragment in the CDNA5 A-matrix layout from LDS.
// (ISA 7.12.2): lanes 0-15 hold row M=lane, K={0..7,16..23};
// lanes 16-31 hold row M=lane-16, K={8..15,24..31}; 2 halves per VGPR.
// The same routine serves B fragments when the tile is stored [N][K].
__device__ __forceinline__ v8u load_frag(const u16* lds, int stride, int row0, int k0) {
  int lane = threadIdx.x & 31;
  const u16* p = lds + (size_t)(row0 + (lane & 15)) * stride + k0 + ((lane >> 4) << 3);
  v8u f;
#pragma unroll
  for (int i = 0; i < 8; i++) {
    int kk = ((i & 4) << 2) + ((i & 3) << 1);  // (i<4?0:16) + (i&3)*2
    f[i] = *(const u32*)(p + kk);
  }
  return f;
}

// ---------------------------------------------------------------- fp32->bf16
__global__ void cvt_f32_bf16(const float* __restrict__ in, u16* __restrict__ out, int n) {
  int i = (blockIdx.x * 256 + threadIdx.x) * 4;
  if (i >= n) return;
  float4 v = *(const float4*)(in + i);
  uint2 o;
  o.x = (u32)f2bf(v.x) | ((u32)f2bf(v.y) << 16);
  o.y = (u32)f2bf(v.z) | ((u32)f2bf(v.w) << 16);
  *(uint2*)(out + i) = o;
}

// ---------------------------------------------------------------- WMMA GEMM
// C[M,N] = A[M,K] * B[K,N], bf16 inputs, f32 accumulate.
// Block tile 128x128, K-step 32, 256 threads = 8 waves (2x4 wave grid),
// each wave does a 64x32 sub-tile = 4x2 WMMA accumulators.
// A tile is DMA'd to LDS by the TDM (wave 0); all waves stage B transposed.
template <int STORE_BF16>
__global__ __launch_bounds__(256) void gemm_bf16_k(const u16* __restrict__ A,
                                                   const u16* __restrict__ B,
                                                   void* __restrict__ Cout,
                                                   int M, int N, int K) {
  constexpr int BM = 128, BN = 128, BK = 32;
  constexpr int AS = BK + 8;  // 40 halves: 32 + pad (pad_interval=16dw, pad=4dw)
  constexpr int BS = BK + 8;
  __shared__ u16 sA[BM * AS];
  __shared__ u16 sB[BN * BS];  // stored transposed: [n][k]
  int tid = threadIdx.x;
  int wave = tid >> 5, lane = tid & 31;
  int wm = (wave & 1) * 64;
  int wn = (wave >> 1) * 32;
  int bm0 = blockIdx.y * BM;
  int bn0 = blockIdx.x * BN;
  u32 sA_off = (u32)(size_t)(const void*)sA;  // flat low 32 bits == LDS offset

  v8f acc[4][2];
#pragma unroll
  for (int a = 0; a < 4; a++)
#pragma unroll
    for (int b = 0; b < 2; b++) acc[a][b] = (v8f){0, 0, 0, 0, 0, 0, 0, 0};

  for (int k0 = 0; k0 < K; k0 += BK) {
    // A tile 128x32: Tensor Data Mover, padded LDS rows (stride 40 halves)
    if (wave == 0) {
      tdm_load_2d_bf16(A + (size_t)bm0 * K + k0, sA_off,
                       /*tdim0*/ (u32)(K - k0), /*tdim1*/ (u32)(M - bm0),
                       /*tile0*/ BK, /*tile1*/ BM, /*stride0*/ (u32)K,
                       /*pad_icode 16dw*/ 3u, /*pad_acode 4dw*/ 3u);
      __builtin_amdgcn_s_wait_tensorcnt(0);
    }
    // B tile 32x128 loaded coalesced, stored transposed [n][k]
#pragma unroll
    for (int i = 0; i < 2; i++) {
      int idx = tid + i * 256;          // 0..511
      int kr = idx >> 4;                // 32 k-rows, 16 chunks of 8
      int c = (idx & 15) << 3;
      uint4 v = *(const uint4*)(B + (size_t)(k0 + kr) * N + bn0 + c);
      const u16* hv = (const u16*)&v;
#pragma unroll
      for (int j = 0; j < 8; j++) sB[(c + j) * BS + kr] = hv[j];
    }
    if (k0 + BK < K)  // hint the next B slab into cache
      __builtin_prefetch(B + (size_t)(k0 + BK + (tid >> 3)) * N + bn0 + ((tid & 7) << 4), 0, 0);
    __syncthreads();

    v8u af[4], bfr[2];
#pragma unroll
    for (int mt = 0; mt < 4; mt++) af[mt] = load_frag(sA, AS, wm + mt * 16, 0);
#pragma unroll
    for (int nt = 0; nt < 2; nt++) bfr[nt] = load_frag(sB, BS, wn + nt * 16, 0);
#pragma unroll
    for (int mt = 0; mt < 4; mt++)
#pragma unroll
      for (int nt = 0; nt < 2; nt++) acc[mt][nt] = wmma_bf16(af[mt], bfr[nt], acc[mt][nt]);
    __syncthreads();
  }

  int half = lane >> 4, nl = lane & 15;
#pragma unroll
  for (int mt = 0; mt < 4; mt++)
#pragma unroll
    for (int nt = 0; nt < 2; nt++) {
      int rowb = bm0 + wm + mt * 16 + half * 8;
      int col = bn0 + wn + nt * 16 + nl;
#pragma unroll
      for (int g = 0; g < 8; g++) {
        float val = acc[mt][nt][g];
        if constexpr (STORE_BF16)
          ((u16*)Cout)[(size_t)(rowb + g) * N + col] = f2bf(val);
        else
          ((float*)Cout)[(size_t)(rowb + g) * N + col] = val;
      }
    }
}

// ---------------------------------------------------------------- RoPE / pack
__global__ void rope_q_k(u16* __restrict__ q, const int* __restrict__ posp) {
  int pos = *posp;
  int i = blockIdx.x * 256 + threadIdx.x;  // over TQ*NH*64
  int j = i & 63;
  int th = i >> 6;
  int h = th & (NH - 1);
  int t = th >> 5;
  u16* base = q + ((size_t)t * NH + h) * DHD;
  float x1 = bf2f(base[j]), x2 = bf2f(base[j + 64]);
  float freq = __powf(ROPE_BASE, -(float)(2 * j) / (float)DHD);
  float th_ = (float)(pos + t) * freq;
  float c, s;
  __sincosf(th_, &s, &c);
  base[j] = f2bf(x1 * c - x2 * s);
  base[j + 64] = f2bf(x1 * s + x2 * c);
}

__global__ void rope_k_k(const u16* __restrict__ kproj, u16* __restrict__ kbf,
                         const int* __restrict__ posp) {
  int pos = *posp;
  int i = blockIdx.x * 256 + threadIdx.x;  // over TQ*NKV*64
  int j = i & 63;
  int th = i >> 6;
  int h = th & (NKV - 1);
  int t = th >> 3;
  const u16* src = kproj + ((size_t)t * NKV + h) * DHD;
  u16* dst = kbf + ((size_t)h * TMAXC + pos + t) * DHD;
  float x1 = bf2f(src[j]), x2 = bf2f(src[j + 64]);
  float freq = __powf(ROPE_BASE, -(float)(2 * j) / (float)DHD);
  float th_ = (float)(pos + t) * freq;
  float c, s;
  __sincosf(th_, &s, &c);
  dst[j] = f2bf(x1 * c - x2 * s);
  dst[j + 64] = f2bf(x1 * s + x2 * c);
}

__global__ void pack_v_k(const u16* __restrict__ vproj, u16* __restrict__ vbf,
                         const int* __restrict__ posp) {
  int pos = *posp;
  int i = blockIdx.x * 256 + threadIdx.x;  // over TQ*NKV*DHD
  int d = i & (DHD - 1);
  int th = i >> 7;
  int h = th & (NKV - 1);
  int t = th >> 3;
  vbf[((size_t)h * TMAXC + pos + t) * DHD + d] = vproj[((size_t)t * NKV + h) * DHD + d];
}

// copies any pre-existing cache region [0,pos) into the bf16 K/V buffers
__global__ void cache_pack_k(const float* __restrict__ kc, const float* __restrict__ vc,
                             u16* __restrict__ kbf, u16* __restrict__ vbf,
                             const int* __restrict__ posp) {
  int pos = *posp;
  int i = blockIdx.x * 256 + threadIdx.x;  // over NKV*TMAXC*DHD
  int t = (i >> 7) & (TMAXC - 1);
  if (t >= pos) return;
  kbf[i] = f2bf(kc[i]);
  vbf[i] = f2bf(vc[i]);
}

// ---------------------------------------------------------------- flash attn
__global__ __launch_bounds__(256) void flash_attn_k(const u16* __restrict__ qbf,
                                                    const u16* __restrict__ kbf,
                                                    const u16* __restrict__ vbf,
                                                    u16* __restrict__ attn,
                                                    const int* __restrict__ posp) {
  constexpr int KS = DHD + 8;  // 136 halves (TDM pad: 64dw interval + 4dw pad)
  constexpr int VS = 64 + 2;   // V^T tile stride (dword-aligned columns)
  constexpr int PS = 64 + 8;   // P tile stride, 144B rows
  __shared__ u16 sK[64 * KS];          // keys row-major: B layout [n=key][k=dh]
  __shared__ u16 sV[DHD * VS];         // V transposed:   B layout [n=dh][k=key]
  __shared__ u16 sP[8 * 16 * PS];      // per-wave P stage (C-layout -> A-layout)

  int pos = *posp;
  int h = blockIdx.y;
  int hk = h >> 2;  // NH/NKV = 4
  int q0 = blockIdx.x * 128;
  int tid = threadIdx.x, wave = tid >> 5, lane = tid & 31;
  int half = lane >> 4, nl = lane & 15;
  int qrow = q0 + wave * 16;
  u32 sK_off = (u32)(size_t)(const void*)sK;

  // preload this wave's Q fragments (16 rows x 128 dh = 4 A-frags)
  v8u qf[4];
  {
    const u16* qp = qbf + ((size_t)(qrow + nl) * NH + h) * DHD + (half << 3);
#pragma unroll
    for (int ks = 0; ks < 4; ks++) {
#pragma unroll
      for (int i = 0; i < 8; i++) {
        int kk = ks * 32 + ((i & 4) << 2) + ((i & 3) << 1);
        qf[ks][i] = *(const u32*)(qp + kk);
      }
    }
  }

  float mrun[8], lrun[8];
  v8f o[8];
#pragma unroll
  for (int g = 0; g < 8; g++) { mrun[g] = -1e30f; lrun[g] = 0.0f; }
#pragma unroll
  for (int nt = 0; nt < 8; nt++) o[nt] = (v8f){0, 0, 0, 0, 0, 0, 0, 0};

  int Tk = pos + TQ;
  int kend = pos + q0 + 128;
  if (kend > Tk) kend = Tk;
  int nkb = (kend + 63) >> 6;
  const float scale = 0.08838834764831845f;  // 1/sqrt(128)

  for (int kb = 0; kb < nkb; kb++) {
    int kbase = kb * 64;
    const u16* Kg = kbf + ((size_t)hk * TMAXC + kbase) * DHD;
    const u16* Vg = vbf + ((size_t)hk * TMAXC + kbase) * DHD;
    // K tile 64x128 via Tensor Data Mover; rows >= Tk are zero-filled by
    // the TDM OOB rule (tensor_dim1 = Tk - kbase).
    if (wave == 0) {
      tdm_load_2d_bf16(Kg, sK_off,
                       /*tdim0*/ DHD, /*tdim1*/ (u32)(Tk - kbase),
                       /*tile0*/ DHD, /*tile1*/ 64, /*stride0*/ DHD,
                       /*pad_icode 64dw*/ 5u, /*pad_acode 4dw*/ 3u);
      __builtin_amdgcn_s_wait_tensorcnt(0);
    }
    // V tile loaded coalesced, stored transposed (all waves)
    {
      uint4 zero4 = {0, 0, 0, 0};
#pragma unroll
      for (int i = 0; i < 4; i++) {
        int idx = tid + i * 256;  // 0..1023 uint4 chunks
        int r = idx >> 4;
        int c = (idx & 15) << 3;
        bool live = (kbase + r) < Tk;
        uint4 vv = live ? *(const uint4*)(Vg + (size_t)r * DHD + c) : zero4;
        const u16* hv = (const u16*)&vv;
#pragma unroll
        for (int j = 0; j < 8; j++) sV[(c + j) * VS + r] = hv[j];
      }
    }
    __syncthreads();

    // S = Q * K^T  (16x64 per wave)
    v8f s[4];
#pragma unroll
    for (int nt = 0; nt < 4; nt++) s[nt] = (v8f){0, 0, 0, 0, 0, 0, 0, 0};
#pragma unroll
    for (int ks = 0; ks < 4; ks++) {
#pragma unroll
      for (int nt = 0; nt < 4; nt++) {
        v8u bfrag = load_frag(sK, KS, nt * 16, ks * 32);
        s[nt] = wmma_bf16(qf[ks], bfrag, s[nt]);
      }
    }

    // causal mask + scale
#pragma unroll
    for (int nt = 0; nt < 4; nt++) {
      int kcol = kbase + nt * 16 + nl;
#pragma unroll
      for (int g = 0; g < 8; g++) {
        int qabs = pos + qrow + half * 8 + g;
        float val = s[nt][g] * scale;
        s[nt][g] = (kcol <= qabs && kcol < Tk) ? val : -1e30f;
      }
    }

    // online softmax (rows live across 16 lanes of a half-wave)
#pragma unroll
    for (int g = 0; g < 8; g++) {
      float v = fmaxf(fmaxf(s[0][g], s[1][g]), fmaxf(s[2][g], s[3][g]));
      v = fmaxf(v, __shfl_xor(v, 1, 32));
      v = fmaxf(v, __shfl_xor(v, 2, 32));
      v = fmaxf(v, __shfl_xor(v, 4, 32));
      v = fmaxf(v, __shfl_xor(v, 8, 32));
      float mn = fmaxf(mrun[g], v);
      float fac = __expf(mrun[g] - mn);
      lrun[g] *= fac;
#pragma unroll
      for (int nt = 0; nt < 8; nt++) o[nt][g] *= fac;
      mrun[g] = mn;
      float ssum = 0.0f;
#pragma unroll
      for (int nt = 0; nt < 4; nt++) {
        float p = __expf(s[nt][g] - mn);
        s[nt][g] = p;
        ssum += p;
      }
      ssum += __shfl_xor(ssum, 1, 32);
      ssum += __shfl_xor(ssum, 2, 32);
      ssum += __shfl_xor(ssum, 4, 32);
      ssum += __shfl_xor(ssum, 8, 32);
      lrun[g] += ssum;
    }

    // stage P (C-layout -> row-major) for A-fragment reload
    u16* Pw = sP + wave * 16 * PS;
#pragma unroll
    for (int nt = 0; nt < 4; nt++)
#pragma unroll
      for (int g = 0; g < 8; g++) Pw[(half * 8 + g) * PS + nt * 16 + nl] = f2bf(s[nt][g]);
    __syncthreads();

    // O += P * V  (16x128 per wave)
#pragma unroll
    for (int ks = 0; ks < 2; ks++) {
      v8u pf = load_frag(Pw, PS, 0, ks * 32);
#pragma unroll
      for (int nt = 0; nt < 8; nt++) {
        v8u vf = load_frag(sV, VS, nt * 16, ks * 32);
        o[nt] = wmma_bf16(pf, vf, o[nt]);
      }
    }
    __syncthreads();
  }

  // normalize + store bf16 attention output [T][NH*DHD]
#pragma unroll
  for (int g = 0; g < 8; g++) {
    float inv = 1.0f / lrun[g];
    int row = qrow + half * 8 + g;
    u16* outp = attn + ((size_t)row * NH + h) * DHD;
#pragma unroll
    for (int nt = 0; nt < 8; nt++) outp[nt * 16 + nl] = f2bf(o[nt][g] * inv);
  }
}

// ---------------------------------------------------------------- launcher
extern "C" void kernel_launch(void* const* d_in, const int* in_sizes, int n_in,
                              void* d_out, int out_size, void* d_ws, size_t ws_size,
                              hipStream_t stream) {
  const float* hs = (const float*)d_in[0];
  const float* Wq = (const float*)d_in[1];
  const float* Wk = (const float*)d_in[2];
  const float* Wv = (const float*)d_in[3];
  const float* Wo = (const float*)d_in[4];
  const float* kc = (const float*)d_in[5];
  const float* vc = (const float*)d_in[6];
  const int* pos = (const int*)d_in[7];

  char* ws = (char*)d_ws;
  size_t off = 0;
  auto take = [&](size_t bytes) -> void* {
    void* p = ws + off;
    off += (bytes + 255) & ~(size_t)255;
    return p;
  };
  u16* hs_bf = (u16*)take((size_t)TQ * HIDW * 2);
  u16* wq_bf = (u16*)take((size_t)HIDW * NH * DHD * 2);
  u16* wk_bf = (u16*)take((size_t)HIDW * NKV * DHD * 2);
  u16* wv_bf = (u16*)take((size_t)HIDW * NKV * DHD * 2);
  u16* wo_bf = (u16*)take((size_t)NH * DHD * HIDW * 2);
  u16* q_bf  = (u16*)take((size_t)TQ * NH * DHD * 2);
  u16* kproj = (u16*)take((size_t)TQ * NKV * DHD * 2);
  u16* vproj = (u16*)take((size_t)TQ * NKV * DHD * 2);
  u16* kbf   = (u16*)take((size_t)NKV * TMAXC * DHD * 2);
  u16* vbf   = (u16*)take((size_t)NKV * TMAXC * DHD * 2);
  u16* attn  = (u16*)take((size_t)TQ * NH * DHD * 2);

  // fp32 -> bf16 conversions
  cvt_f32_bf16<<<(TQ * HIDW) / 1024, 256, 0, stream>>>(hs, hs_bf, TQ * HIDW);
  cvt_f32_bf16<<<(HIDW * NH * DHD) / 1024, 256, 0, stream>>>(Wq, wq_bf, HIDW * NH * DHD);
  cvt_f32_bf16<<<(HIDW * NKV * DHD) / 1024, 256, 0, stream>>>(Wk, wk_bf, HIDW * NKV * DHD);
  cvt_f32_bf16<<<(HIDW * NKV * DHD) / 1024, 256, 0, stream>>>(Wv, wv_bf, HIDW * NKV * DHD);
  cvt_f32_bf16<<<(NH * DHD * HIDW) / 1024, 256, 0, stream>>>(Wo, wo_bf, NH * DHD * HIDW);

  // projections (bf16 out)
  gemm_bf16_k<1><<<dim3((NH * DHD) / 128, TQ / 128), 256, 0, stream>>>(
      hs_bf, wq_bf, q_bf, TQ, NH * DHD, HIDW);
  gemm_bf16_k<1><<<dim3((NKV * DHD) / 128, TQ / 128), 256, 0, stream>>>(
      hs_bf, wk_bf, kproj, TQ, NKV * DHD, HIDW);
  gemm_bf16_k<1><<<dim3((NKV * DHD) / 128, TQ / 128), 256, 0, stream>>>(
      hs_bf, wv_bf, vproj, TQ, NKV * DHD, HIDW);

  // RoPE + head-major packing + cache region
  rope_q_k<<<(TQ * NH * 64) / 256, 256, 0, stream>>>(q_bf, pos);
  rope_k_k<<<(TQ * NKV * 64) / 256, 256, 0, stream>>>(kproj, kbf, pos);
  pack_v_k<<<(TQ * NKV * DHD) / 256, 256, 0, stream>>>(vproj, vbf, pos);
  cache_pack_k<<<(NKV * TMAXC * DHD) / 256, 256, 0, stream>>>(kc, vc, kbf, vbf, pos);

  // causal flash attention
  flash_attn_k<<<dim3(TQ / 128, NH), 256, 0, stream>>>(q_bf, kbf, vbf, attn, pos);

  // output projection (f32 out)
  gemm_bf16_k<0><<<dim3(HIDW / 128, TQ / 128), 256, 0, stream>>>(
      attn, wo_bf, d_out, TQ, HIDW, NH * DHD);
}
